// BailingMoeBlock_87333864996962
// MI455X (gfx1250) — matrix operand order repaired and checked by reading, not compiled, not run on hardware
//
#include <hip/hip_runtime.h>
#include <hip/hip_bf16.h>
#include <stdint.h>

// ---------------- problem constants ----------------
#define T_TOK 2048
#define HID   1024
#define NH    8
#define NKV   2
#define HD    128
#define NE    16
#define TOPK  4
#define NG    4
#define FF    1024
#define FSH   1024
#define QKVN  ((NH + 2 * NKV) * HD)            // 1536
#define MAXROWS (T_TOK * TOPK + NE * 128)      // 10240 padded rows
#define MAXMT   (MAXROWS / 128)                // 80 m-tiles

// ---------------- WMMA plumbing ----------------
typedef __attribute__((ext_vector_type(16))) __bf16 v16bf;
typedef __attribute__((ext_vector_type(8)))  float  v8f;

union Frag {
  unsigned int u[8];
  v16bf v;
};

__device__ inline unsigned short f2bf(float f) {
  unsigned int u = __float_as_uint(f);
  u += 0x7FFFu + ((u >> 16) & 1u);   // round-to-nearest-even
  return (unsigned short)(u >> 16);
}

__device__ inline v8f wmma_bf16(const Frag& a, const Frag& b, v8f c) {
  // 8 args: (neg_a, A, neg_b, B, c_mod, C, reuse_a, reuse_b)
  return __builtin_amdgcn_wmma_f32_16x16x32_bf16(false, a.v, false, b.v,
                                                 (short)0, c, false, false);
}

__device__ inline v8f v8f_zero() {
  v8f z;
#pragma unroll
  for (int i = 0; i < 8; i++) z[i] = 0.0f;
  return z;
}

// 16B async copy global -> LDS (gfx1250 GLOBAL_LOAD_ASYNC_TO_LDS_B128, ASYNCcnt)
__device__ inline void async_g2l_b128(void* lds_dst, const void* gsrc) {
  unsigned int laddr = (unsigned int)(size_t)lds_dst;            // LDS offset
  unsigned long long gaddr = (unsigned long long)(size_t)gsrc;   // 64-bit VA
  asm volatile("global_load_async_to_lds_b128 %0, %1, off"
               :
               : "v"(laddr), "v"(gaddr)
               : "memory");
}

__device__ inline void wait_async0() {
  asm volatile("s_wait_asynccnt 0x0" ::: "memory");
}

// ---------------- f32 -> bf16 bulk convert ----------------
__global__ __launch_bounds__(256) void k_f2bf(const float* __restrict__ s,
                                              unsigned short* __restrict__ d,
                                              long long n) {
  long long i = (long long)blockIdx.x * blockDim.x + threadIdx.x;
  long long stride = (long long)gridDim.x * blockDim.x;
  for (; i < n; i += stride) d[i] = f2bf(s[i]);
}

// ---------------- fused (x+add) + RMSNorm ----------------
__global__ __launch_bounds__(256) void k_rmsnorm(
    const float* __restrict__ x, const float* __restrict__ add,
    const float* __restrict__ w, float* __restrict__ resid_out,
    unsigned short* __restrict__ out_bf, float* __restrict__ out_f32) {
  int t = blockIdx.x;
  int tid = threadIdx.x;
  float4 xv = *(const float4*)(x + (long long)t * HID + tid * 4);
  if (add) {
    float4 av = *(const float4*)(add + (long long)t * HID + tid * 4);
    xv.x += av.x; xv.y += av.y; xv.z += av.z; xv.w += av.w;
  }
  if (resid_out) *(float4*)(resid_out + (long long)t * HID + tid * 4) = xv;
  float ss = xv.x * xv.x + xv.y * xv.y + xv.z * xv.z + xv.w * xv.w;
#pragma unroll
  for (int o = 1; o < 32; o <<= 1) ss += __shfl_xor(ss, o, 32);
  __shared__ float red[8];
  int wid = tid >> 5, lane = tid & 31;
  if (lane == 0) red[wid] = ss;
  __syncthreads();
  float tot = 0.0f;
#pragma unroll
  for (int i = 0; i < 8; i++) tot += red[i];
  float rms = rsqrtf(tot / (float)HID + 1e-6f);
  float4 wv = *(const float4*)(w + tid * 4);
  float o0 = xv.x * rms * wv.x, o1 = xv.y * rms * wv.y;
  float o2 = xv.z * rms * wv.z, o3 = xv.w * rms * wv.w;
  long long base = (long long)t * HID + tid * 4;
  out_bf[base + 0] = f2bf(o0); out_bf[base + 1] = f2bf(o1);
  out_bf[base + 2] = f2bf(o2); out_bf[base + 3] = f2bf(o3);
  if (out_f32) {
    out_f32[base + 0] = o0; out_f32[base + 1] = o1;
    out_f32[base + 2] = o2; out_f32[base + 3] = o3;
  }
}

// ---------------- QK RMSNorm + RoPE + bf16 pack ----------------
__global__ __launch_bounds__(128) void k_qkv_rope(
    const float* __restrict__ qkv, const int* __restrict__ pos_ids,
    const float* __restrict__ qnw, const float* __restrict__ knw,
    unsigned short* __restrict__ q_bf, unsigned short* __restrict__ k_bf,
    unsigned short* __restrict__ v_bf) {
  int t = blockIdx.x;
  int wid = threadIdx.x >> 5, lane = threadIdx.x & 31;
  float pos = (float)pos_ids[t];
  const float* row = qkv + (long long)t * QKVN;
  for (int u = wid; u < 12; u += 4) {   // 8 q-heads, 2 k-heads, 2 v-heads
    float4 xv = *(const float4*)(row + u * HD + lane * 4);
    if (u < 10) {
      float ss = xv.x * xv.x + xv.y * xv.y + xv.z * xv.z + xv.w * xv.w;
#pragma unroll
      for (int o = 1; o < 32; o <<= 1) ss += __shfl_xor(ss, o, 32);
      float rms = rsqrtf(ss / (float)HD + 1e-6f);
      const float* wn = (u < 8) ? qnw : knw;
      float4 wv = *(const float4*)(wn + lane * 4);
      float nv[4] = {xv.x * rms * wv.x, xv.y * rms * wv.y,
                     xv.z * rms * wv.z, xv.w * rms * wv.w};
      unsigned short ob[4];
#pragma unroll
      for (int j = 0; j < 4; j++) {
        float other = __shfl_xor(nv[j], 16, 32);   // partner half of the head dim
        int d = lane * 4 + j;
        int i = (d < 64) ? d : d - 64;
        float freq = __powf(10000.0f, -(float)(2 * i) / (float)HD);
        float ang = pos * freq;
        float c = __cosf(ang), s = __sinf(ang);
        float res = (d < 64) ? (nv[j] * c - other * s) : (nv[j] * c + other * s);
        ob[j] = f2bf(res);
      }
      unsigned short* dst = (u < 8)
          ? q_bf + ((long long)t * NH + u) * HD + lane * 4
          : k_bf + ((long long)t * NKV + (u - 8)) * HD + lane * 4;
#pragma unroll
      for (int j = 0; j < 4; j++) dst[j] = ob[j];
    } else {
      unsigned short* dst = v_bf + ((long long)t * NKV + (u - 10)) * HD + lane * 4;
      dst[0] = f2bf(xv.x); dst[1] = f2bf(xv.y);
      dst[2] = f2bf(xv.z); dst[3] = f2bf(xv.w);
    }
  }
}

// ---------------- flash-style causal GQA attention (WMMA) ----------------
#define AW 2   // waves per block (per-wave LDS tiles, no cross-wave sync needed)
__global__ __launch_bounds__(64) void k_attn(
    const unsigned short* __restrict__ q_bf, const unsigned short* __restrict__ k_bf,
    const unsigned short* __restrict__ v_bf, unsigned short* __restrict__ ctx_bf) {
  __shared__ unsigned short Ks[AW][32][HD];   // [key][d]
  __shared__ unsigned short Vt[AW][HD][32];   // [d][key] (transposed for B-frags)
  __shared__ unsigned short Ps[AW][16][32];   // softmax probs -> A-frag relayout
  int wv = threadIdx.x >> 5, lane = threadIdx.x & 31;
  int task = blockIdx.x * AW + wv;            // NH * (T/16) = 1024 tasks
  int head = task >> 7;                       // T/16 = 128
  int qb = task & 127;
  int q0 = qb * 16;
  int kvh = head / (NH / NKV);
  int lh = lane >> 4, ln = lane & 15;

  // Q A-fragments straight from global (row = lane%16, packed K pairs)
  Frag qa[4];
#pragma unroll
  for (int kt = 0; kt < 4; kt++)
#pragma unroll
    for (int v = 0; v < 8; v++) {
      int k = kt * 32 + 2 * v + 16 * lh;
      qa[kt].u[v] =
          *(const unsigned int*)&q_bf[((long long)(q0 + ln) * NH + head) * HD + k];
    }

  v8f ctx[8];
  float mrow[8], lrow[8];
#pragma unroll
  for (int i = 0; i < 8; i++) { ctx[i] = v8f_zero(); mrow[i] = -1e30f; lrow[i] = 0.0f; }

  int nkb = (q0 + 47) >> 5;   // 32-key blocks covering keys <= q0+15 (T mult of 32)
  for (int kb = 0; kb < nkb; kb++) {
    // K tile: async global->LDS (row-major, no reshape). V tile: manual transpose.
    for (int c = lane; c < 32 * (HD / 8); c += 32) {
      int key = c >> 4, kc = (c & 15) * 8;
      long long src = ((long long)(kb * 32 + key) * NKV + kvh) * HD + kc;
      async_g2l_b128(&Ks[wv][key][kc], &k_bf[src]);
      uint4 vd = *(const uint4*)&v_bf[src];
      const unsigned short* vs = (const unsigned short*)&vd;
#pragma unroll
      for (int j = 0; j < 8; j++) Vt[wv][kc + j][key] = vs[j];
    }
    wait_async0();                   // ASYNCcnt==0: K tile resident in LDS
    asm volatile("" ::: "memory");   // per-wave DS ops are in-order

    // S = Q * K^T  (two 16x16 C tiles over 32 keys)
    v8f s0 = v8f_zero(), s1 = v8f_zero();
#pragma unroll
    for (int kt = 0; kt < 4; kt++) {
      Frag b0, b1;
#pragma unroll
      for (int v = 0; v < 8; v++) {
        int d = kt * 32 + 2 * v + 16 * lh;
        b0.u[v] = *(const unsigned int*)&Ks[wv][ln][d];
        b1.u[v] = *(const unsigned int*)&Ks[wv][16 + ln][d];
      }
      s0 = wmma_bf16(qa[kt], b0, s0);
      s1 = wmma_bf16(qa[kt], b1, s1);
    }

    const float scale = 0.088388347648318447f;   // 1/sqrt(128)
    int c0 = kb * 32 + ln, c1 = c0 + 16;
#pragma unroll
    for (int v = 0; v < 8; v++) {
      int r = q0 + v + 8 * lh;
      float a = (c0 <= r) ? s0[v] * scale : -1e30f;
      float b = (c1 <= r) ? s1[v] * scale : -1e30f;
      float mx = fmaxf(a, b);
#pragma unroll
      for (int o = 1; o < 16; o <<= 1) mx = fmaxf(mx, __shfl_xor(mx, o, 32));
      float mnew = fmaxf(mrow[v], mx);
      float cf = __expf(mrow[v] - mnew);
      float pa = __expf(a - mnew), pb = __expf(b - mnew);
      float ps = pa + pb;
#pragma unroll
      for (int o = 1; o < 16; o <<= 1) ps += __shfl_xor(ps, o, 32);
      lrow[v] = lrow[v] * cf + ps;
      mrow[v] = mnew;
#pragma unroll
      for (int nt = 0; nt < 8; nt++) ctx[nt][v] *= cf;
      Ps[wv][v + 8 * lh][ln] = f2bf(pa);
      Ps[wv][v + 8 * lh][16 + ln] = f2bf(pb);
    }
    asm volatile("" ::: "memory");

    // ctx += P * V
    Frag pf;
#pragma unroll
    for (int v = 0; v < 8; v++) {
      int kp = 2 * v + 16 * lh;
      pf.u[v] = *(const unsigned int*)&Ps[wv][ln][kp];
    }
#pragma unroll
    for (int nt = 0; nt < 8; nt++) {
      Frag bv;
#pragma unroll
      for (int v = 0; v < 8; v++)
        bv.u[v] = *(const unsigned int*)&Vt[wv][nt * 16 + ln][2 * v + 16 * lh];
      ctx[nt] = wmma_bf16(pf, bv, ctx[nt]);
    }
  }

#pragma unroll
  for (int v = 0; v < 8; v++) lrow[v] = 1.0f / lrow[v];
#pragma unroll
  for (int nt = 0; nt < 8; nt++)
#pragma unroll
    for (int v = 0; v < 8; v++) {
      float val = ctx[nt][v] * lrow[v];
      ctx_bf[((long long)(q0 + v + 8 * lh) * NH + head) * HD + nt * 16 + ln] =
          f2bf(val);
    }
}

// ---------------- generic WMMA GEMM (gather / grouped / fused-SiLU) ----------------
// C[M,N] = A[M,K](bf16) x B[K,N](bf16). mode 0: f32 out. mode 1: gate-up SiLU:
// accumulates cols [n] and [n+fusedOff] of B, writes bf16 silu(g)*u with ldc=N.
__global__ __launch_bounds__(256) void k_gemm(
    const unsigned short* __restrict__ A, const unsigned short* __restrict__ B,
    void* __restrict__ Cout, int M, int N, int K, int lda, int ldb, int ldc,
    const int* __restrict__ gather, const int* __restrict__ mtileExpert,
    long long expertStride, const int* __restrict__ Mdev, int mode, int fusedOff) {
  __shared__ unsigned short As[128][32];
  __shared__ unsigned short Btg[128][32];   // B transposed: [n][k]
  __shared__ unsigned short Btu[128][32];
  int Mr = Mdev ? *Mdev : M;
  int m0 = blockIdx.y * 128;
  if (m0 >= Mr) return;
  int n0 = blockIdx.x * 128;
  const unsigned short* Bp = B;
  if (mtileExpert) Bp += (long long)mtileExpert[blockIdx.y] * expertStride;
  int tid = threadIdx.x, wid = tid >> 5, lane = tid & 31;
  int lh = lane >> 4, ln = lane & 15;
  int wm = wid >> 1, wn = wid & 1;   // 4x2 wave grid: 32x64 per wave
  v8f acc[2][4], accU[2][4];
#pragma unroll
  for (int mi = 0; mi < 2; mi++)
#pragma unroll
    for (int ni = 0; ni < 4; ni++) { acc[mi][ni] = v8f_zero(); accU[mi][ni] = v8f_zero(); }

  for (int kk = 0; kk < K; kk += 32) {
    __syncthreads();
    if (gather == nullptr) {
      // dense A rows: async global->LDS copy (no reshape, no VGPR staging)
      for (int c = tid; c < 128 * 32 / 8; c += 256) {
        int r = c >> 2, kc = (c & 3) * 8;
        async_g2l_b128(&As[r][kc], A + (long long)(m0 + r) * lda + kk + kc);
      }
    } else {
      for (int c = tid; c < 128 * 32 / 8; c += 256) {
        int r = c >> 2, kc = (c & 3) * 8;
        int gr = gather[m0 + r];
        uint4 val = {0u, 0u, 0u, 0u};
        if (gr >= 0) val = *(const uint4*)(A + (long long)gr * lda + kk + kc);
        *(uint4*)&As[r][kc] = val;
      }
    }
    for (int c = tid; c < 32 * 128 / 8; c += 256) {   // B tile(s), transposed store
      int kr = c >> 4, nc = (c & 15) * 8;
      const unsigned short* bsrc = Bp + (long long)(kk + kr) * ldb + n0 + nc;
      uint4 val = *(const uint4*)bsrc;
      const unsigned short* vs = (const unsigned short*)&val;
#pragma unroll
      for (int j = 0; j < 8; j++) Btg[nc + j][kr] = vs[j];
      if (mode == 1) {
        uint4 v2 = *(const uint4*)(bsrc + fusedOff);
        const unsigned short* vs2 = (const unsigned short*)&v2;
#pragma unroll
        for (int j = 0; j < 8; j++) Btu[nc + j][kr] = vs2[j];
      }
      if (kk + 32 < K) __builtin_prefetch(bsrc + 32 * ldb, 0, 0);  // global_prefetch_b8
    }
    if (gather == nullptr) wait_async0();   // ASYNCcnt==0 before the barrier
    __syncthreads();

    Frag af[2];
#pragma unroll
    for (int mi = 0; mi < 2; mi++)
#pragma unroll
      for (int v = 0; v < 8; v++) {
        int k = 2 * v + 16 * lh;
        af[mi].u[v] = *(const unsigned int*)&As[wm * 32 + mi * 16 + ln][k];
      }
#pragma unroll
    for (int ni = 0; ni < 4; ni++) {
      int col = wn * 64 + ni * 16 + ln;
      Frag bf;
#pragma unroll
      for (int v = 0; v < 8; v++)
        bf.u[v] = *(const unsigned int*)&Btg[col][2 * v + 16 * lh];
#pragma unroll
      for (int mi = 0; mi < 2; mi++) acc[mi][ni] = wmma_bf16(af[mi], bf, acc[mi][ni]);
      if (mode == 1) {
        Frag bu;
#pragma unroll
        for (int v = 0; v < 8; v++)
          bu.u[v] = *(const unsigned int*)&Btu[col][2 * v + 16 * lh];
#pragma unroll
        for (int mi = 0; mi < 2; mi++) accU[mi][ni] = wmma_bf16(af[mi], bu, accU[mi][ni]);
      }
    }
  }

#pragma unroll
  for (int mi = 0; mi < 2; mi++)
#pragma unroll
    for (int ni = 0; ni < 4; ni++)
#pragma unroll
      for (int v = 0; v < 8; v++) {
        long long m = m0 + wm * 32 + mi * 16 + v + 8 * lh;
        long long n = n0 + wn * 64 + ni * 16 + ln;
        if (mode == 0) {
          ((float*)Cout)[m * ldc + n] = acc[mi][ni][v];
        } else {
          float g = acc[mi][ni][v], u = accU[mi][ni][v];
          float a = g / (1.0f + __expf(-g)) * u;   // silu(g) * u
          ((unsigned short*)Cout)[m * ldc + n] = f2bf(a);
        }
      }
}

// ---------------- MoE routing ----------------
__global__ __launch_bounds__(32) void k_route(
    const float* __restrict__ h2, const float* __restrict__ gate_w,
    const float* __restrict__ bias, int* __restrict__ topk_idx,
    float* __restrict__ topk_w, int* __restrict__ counts) {
  int t = blockIdx.x, lane = threadIdx.x;
  float logit = 0.0f;
  if (lane < NE) {
    const float* hr = h2 + (long long)t * HID;
    for (int h = 0; h < HID; h++) logit += hr[h] * gate_w[h * NE + lane];
  }
  float sv = 1.0f / (1.0f + __expf(-logit));
  float sfc = sv + ((lane < NE) ? bias[lane] : 0.0f);
  float sf[NE], ssv[NE];
  for (int e = 0; e < NE; e++) {
    float a = __shfl(sfc, e, 32);
    float b = __shfl(sv, e, 32);
    if (lane == 0) { sf[e] = a; ssv[e] = b; }
  }
  if (lane == 0) {
    float gsc[NG];
    for (int g = 0; g < NG; g++) {   // top-2 sum per group
      float m1 = -1e30f, m2 = -1e30f;
      for (int j = 0; j < NE / NG; j++) {
        float x = sf[g * (NE / NG) + j];
        if (x > m1) { m2 = m1; m1 = x; } else if (x > m2) m2 = x;
      }
      gsc[g] = m1 + m2;
    }
    int g1 = 0;
    for (int g = 1; g < NG; g++) if (gsc[g] > gsc[g1]) g1 = g;
    int g2 = -1;
    for (int g = 0; g < NG; g++) if (g != g1 && (g2 < 0 || gsc[g] > gsc[g2])) g2 = g;
    bool used[NE];
    for (int e = 0; e < NE; e++) used[e] = false;
    int idx[TOPK]; float wk[TOPK]; float wsum = 0.0f;
    for (int k = 0; k < TOPK; k++) {
      int best = -1;
      for (int e = 0; e < NE; e++) {
        int g = e / (NE / NG);
        if ((g == g1 || g == g2) && !used[e] && (best < 0 || sf[e] > sf[best])) best = e;
      }
      used[best] = true;
      idx[k] = best;
      wk[k] = ssv[best];
      wsum += wk[k];
    }
    float inv = 1.0f / (wsum + 1e-20f);
    for (int k = 0; k < TOPK; k++) {
      topk_idx[t * TOPK + k] = idx[k];
      topk_w[t * TOPK + k] = wk[k] * inv;
      atomicAdd(&counts[idx[k]], 1);
    }
  }
}

__global__ void k_zero_i32(int* p, int n) {
  int i = blockIdx.x * blockDim.x + threadIdx.x;
  if (i < n) p[i] = 0;
}

__global__ __launch_bounds__(256) void k_offsets(
    const int* __restrict__ counts, int* __restrict__ off, int* __restrict__ Mdev,
    int* __restrict__ mtileExp, int* __restrict__ cursor, int* __restrict__ gather) {
  __shared__ int soff[NE + 1];
  if (threadIdx.x == 0) {
    int run = 0;
    for (int e = 0; e < NE; e++) { soff[e] = run; run += (counts[e] + 127) & ~127; }
    soff[NE] = run;
    *Mdev = run;
    for (int e = 0; e <= NE; e++) off[e] = soff[e];
  }
  __syncthreads();
  int tid = threadIdx.x;
  if (tid < NE) cursor[tid] = 0;
  if (tid < MAXMT) {
    int r = tid * 128, e = 0;
    for (int j = 0; j < NE; j++) if (soff[j] <= r) e = j;
    mtileExp[tid] = e;
  }
  for (int i = tid; i < MAXROWS; i += blockDim.x) gather[i] = -1;
}

__global__ __launch_bounds__(256) void k_scatter(
    const int* __restrict__ topk_idx, const int* __restrict__ off,
    int* __restrict__ cursor, int* __restrict__ gather, int* __restrict__ rowOf) {
  int i = blockIdx.x * blockDim.x + threadIdx.x;
  if (i >= T_TOK * TOPK) return;
  int e = topk_idx[i];
  int slot = atomicAdd(&cursor[e], 1);
  int row = off[e] + slot;
  gather[row] = i / TOPK;
  rowOf[i] = row;
}

__global__ __launch_bounds__(256) void k_combine(
    const float* __restrict__ sharedbuf, const float* __restrict__ y,
    const int* __restrict__ rowOf, const float* __restrict__ topk_w,
    float* __restrict__ out) {
  int t = blockIdx.x;
  int col = blockIdx.y * 256 + threadIdx.x;
  float acc = sharedbuf[(long long)t * HID + col];
  for (int k = 0; k < TOPK; k++) {
    int row = rowOf[t * TOPK + k];
    acc += 2.0f * topk_w[t * TOPK + k] * y[(long long)row * HID + col];
  }
  out[(long long)t * HID + col] = acc;
}

// ---------------- host orchestration ----------------
extern "C" void kernel_launch(void* const* d_in, const int* in_sizes, int n_in,
                              void* d_out, int out_size, void* d_ws, size_t ws_size,
                              hipStream_t stream) {
  (void)in_sizes; (void)n_in; (void)out_size; (void)ws_size;
  const float* hs    = (const float*)d_in[0];
  const float* resd  = (const float*)d_in[1];
  const int*   pos   = (const int*)d_in[2];
  const float* ln1w  = (const float*)d_in[3];
  const float* qkvw  = (const float*)d_in[4];
  const float* qnw   = (const float*)d_in[5];
  const float* knw   = (const float*)d_in[6];
  const float* dnsw  = (const float*)d_in[7];
  const float* ln2w  = (const float*)d_in[8];
  const float* gatew = (const float*)d_in[9];
  const float* ebias = (const float*)d_in[10];
  const float* w13   = (const float*)d_in[11];
  const float* w2    = (const float*)d_in[12];
  const float* wsgu  = (const float*)d_in[13];
  const float* wsdn  = (const float*)d_in[14];
  float* outp = (float*)d_out;

  char* base = (char*)d_ws;
  size_t off = 0;
  auto alloc = [&](size_t bytes) -> char* {
    char* p = base + off;
    off = (off + bytes + 255) & ~(size_t)255;
    return p;
  };
  // Phase-1 region (dead before the MoE down-GEMM): reused as yBuf (42MB < 44MB).
  float*          residW  = (float*)alloc((size_t)T_TOK * HID * 4);
  unsigned short* h_bf    = (unsigned short*)alloc((size_t)T_TOK * HID * 2);
  float*          qkv     = (float*)alloc((size_t)T_TOK * QKVN * 4);
  unsigned short* q_bf    = (unsigned short*)alloc((size_t)T_TOK * NH * HD * 2);
  unsigned short* k_bf    = (unsigned short*)alloc((size_t)T_TOK * NKV * HD * 2);
  unsigned short* v_bf    = (unsigned short*)alloc((size_t)T_TOK * NKV * HD * 2);
  unsigned short* ctx_bf  = (unsigned short*)alloc((size_t)T_TOK * NH * HD * 2);
  float*          attn_o  = (float*)alloc((size_t)T_TOK * HID * 4);
  float*          yBuf    = (float*)base;   // alias over phase-1 region
  float*          h2f     = (float*)alloc((size_t)T_TOK * HID * 4);
  unsigned short* h2bf    = (unsigned short*)alloc((size_t)T_TOK * HID * 2);
  unsigned short* wqkv_bf = (unsigned short*)alloc((size_t)HID * QKVN * 2);
  unsigned short* wdns_bf = (unsigned short*)alloc((size_t)NH * HD * HID * 2);
  unsigned short* w13_bf  = (unsigned short*)alloc((size_t)NE * HID * 2 * FF * 2);
  unsigned short* w2_bf   = (unsigned short*)alloc((size_t)NE * FF * HID * 2);
  unsigned short* wsgu_bf = (unsigned short*)alloc((size_t)HID * 2 * FSH * 2);
  unsigned short* wsdn_bf = (unsigned short*)alloc((size_t)FSH * HID * 2);
  int*   topk_idx = (int*)alloc((size_t)T_TOK * TOPK * 4);
  float* topk_w   = (float*)alloc((size_t)T_TOK * TOPK * 4);
  int*   counts   = (int*)alloc(64);
  int*   cursor   = (int*)alloc(64);
  int*   offs     = (int*)alloc((NE + 1) * 4);
  int*   Mdev     = (int*)alloc(4);
  int*   mtileExp = (int*)alloc(MAXMT * 4);
  int*   gather   = (int*)alloc((size_t)MAXROWS * 4);
  int*   rowOf    = (int*)alloc((size_t)T_TOK * TOPK * 4);
  unsigned short* act_bf  = (unsigned short*)alloc((size_t)MAXROWS * FF * 2);
  unsigned short* sact_bf = (unsigned short*)alloc((size_t)T_TOK * FSH * 2);
  float*          sharedb = (float*)alloc((size_t)T_TOK * HID * 4);

  // Weight conversion f32 -> bf16 (one bandwidth pass, halves GEMM traffic)
  k_f2bf<<<1024, 256, 0, stream>>>(qkvw, wqkv_bf, (long long)HID * QKVN);
  k_f2bf<<<1024, 256, 0, stream>>>(dnsw, wdns_bf, (long long)NH * HD * HID);
  k_f2bf<<<1024, 256, 0, stream>>>(w13, w13_bf, (long long)NE * HID * 2 * FF);
  k_f2bf<<<1024, 256, 0, stream>>>(w2, w2_bf, (long long)NE * FF * HID);
  k_f2bf<<<1024, 256, 0, stream>>>(wsgu, wsgu_bf, (long long)HID * 2 * FSH);
  k_f2bf<<<1024, 256, 0, stream>>>(wsdn, wsdn_bf, (long long)FSH * HID);

  // Attention path
  k_rmsnorm<<<T_TOK, 256, 0, stream>>>(hs, resd, ln1w, residW, h_bf, nullptr);
  k_gemm<<<dim3(QKVN / 128, T_TOK / 128), 256, 0, stream>>>(
      h_bf, wqkv_bf, qkv, T_TOK, QKVN, HID, HID, QKVN, QKVN,
      nullptr, nullptr, 0, nullptr, 0, 0);
  k_qkv_rope<<<T_TOK, 128, 0, stream>>>(qkv, pos, qnw, knw, q_bf, k_bf, v_bf);
  k_attn<<<(NH * (T_TOK / 16)) / AW, 32 * AW, 0, stream>>>(q_bf, k_bf, v_bf, ctx_bf);
  k_gemm<<<dim3(HID / 128, T_TOK / 128), 256, 0, stream>>>(
      ctx_bf, wdns_bf, attn_o, T_TOK, HID, NH * HD, NH * HD, HID, HID,
      nullptr, nullptr, 0, nullptr, 0, 0);
  // resid2 goes straight to d_out second half
  k_rmsnorm<<<T_TOK, 256, 0, stream>>>(attn_o, residW, ln2w,
                                       outp + (size_t)T_TOK * HID, h2bf, h2f);

  // Routing + dispatch
  k_zero_i32<<<1, 32, 0, stream>>>(counts, NE);
  k_route<<<T_TOK, 32, 0, stream>>>(h2f, gatew, ebias, topk_idx, topk_w, counts);
  k_offsets<<<1, 256, 0, stream>>>(counts, offs, Mdev, mtileExp, cursor, gather);
  k_scatter<<<(T_TOK * TOPK) / 256, 256, 0, stream>>>(topk_idx, offs, cursor,
                                                      gather, rowOf);

  // Shared expert (fused SiLU gate-up, then down-proj)
  k_gemm<<<dim3(FSH / 128, T_TOK / 128), 256, 0, stream>>>(
      h2bf, wsgu_bf, sact_bf, T_TOK, FSH, HID, HID, 2 * FSH, FSH,
      nullptr, nullptr, 0, nullptr, 1, FSH);
  k_gemm<<<dim3(HID / 128, T_TOK / 128), 256, 0, stream>>>(
      sact_bf, wsdn_bf, sharedb, T_TOK, HID, FSH, FSH, HID, HID,
      nullptr, nullptr, 0, nullptr, 0, 0);

  // Routed experts: gathered fused gate-up GEMM, then grouped down GEMM
  k_gemm<<<dim3(FF / 128, MAXMT), 256, 0, stream>>>(
      h2bf, w13_bf, act_bf, MAXROWS, FF, HID, HID, 2 * FF, FF,
      gather, mtileExp, (long long)HID * 2 * FF, Mdev, 1, FF);
  k_gemm<<<dim3(HID / 128, MAXMT), 256, 0, stream>>>(
      act_bf, w2_bf, yBuf, MAXROWS, HID, FF, FF, HID, HID,
      nullptr, mtileExp, (long long)FF * HID, Mdev, 0, 0);

  // Combine: shared + 2.0 * sum_k w_k * y[row]
  k_combine<<<dim3(T_TOK, HID / 256), 256, 0, stream>>>(sharedb, yBuf, rowOf,
                                                        topk_w, outp);
}
